// MambaBackbone_51539608397
// MI455X (gfx1250) — matrix operand order repaired and checked by reading, not compiled, not run on hardware
//
#include <hip/hip_runtime.h>
#include <hip/hip_bf16.h>
#include <math.h>

// ---------------- problem constants (from reference) ----------------
#define B_    4
#define T_    2048
#define DM    512     // D_MODEL
#define DST   64      // D_STATE
#define DCONV 4
#define DIN   1024    // D_INNER
#define NH    16      // NHEADS
#define HD    64      // HEADDIM
#define DXBC  1152    // D_INNER + 2*D_STATE
#define DPROJ 2192    // 2*D_INNER + 2*D_STATE + NHEADS
#define NC    8       // T / CHUNK
#define CHK   256
#define XPAD  16      // pad sXT rows to 272 bf16 = 544B (32B-aligned rows)

typedef __attribute__((ext_vector_type(16))) __bf16 v16bf;
typedef __attribute__((ext_vector_type(8)))  __bf16 v8bf;
typedef __attribute__((ext_vector_type(8)))  float  v8f;

#define WMMA_BF16(A, Bf, Cf) \
  __builtin_amdgcn_wmma_f32_16x16x32_bf16(false, (A), false, (Bf), (short)0, (Cf), false, false)

// ---------------- fragment load helpers ----------------
// A-fragment (16x32 bf16, ISA 7.12.2): lane (m = lane&15, half = lane>>4),
// element e -> k = half*16 + e (this file's order, not the ISA table order):
// one contiguous 16-run. B-fragment: k = half*16 + e: one contiguous 16-run.

__device__ __forceinline__ v16bf cvt_runs(const float* r0, const float* r1) {
  float t[16];
  *(float4*)(t + 0)  = ((const float4*)r0)[0];
  *(float4*)(t + 4)  = ((const float4*)r0)[1];
  *(float4*)(t + 8)  = ((const float4*)r1)[0];
  *(float4*)(t + 12) = ((const float4*)r1)[1];
  v16bf f;
#pragma unroll
  for (int e = 0; e < 16; ++e) f[e] = (__bf16)t[e];
  return f;
}

__device__ __forceinline__ v16bf cvt16(const float* p) {
  float t[16];
  *(float4*)(t + 0)  = ((const float4*)p)[0];
  *(float4*)(t + 4)  = ((const float4*)p)[1];
  *(float4*)(t + 8)  = ((const float4*)p)[2];
  *(float4*)(t + 12) = ((const float4*)p)[3];
  v16bf f;
#pragma unroll
  for (int e = 0; e < 16; ++e) f[e] = (__bf16)t[e];
  return f;
}

__device__ __forceinline__ v16bf lds_runs(const __bf16* r0, const __bf16* r1) {
  v8bf h0 = *(const v8bf*)r0;   // 16B ds_load_b128
  v8bf h1 = *(const v8bf*)r1;
  return __builtin_shufflevector(h0, h1, 0,1,2,3,4,5,6,7,8,9,10,11,12,13,14,15);
}

// =====================================================================
// 1/8. WMMA GEMM: C[M,N] = A[M,K] @ B[K,N].  Requires M%64==0, K%32==0, N%8==0.
// LDS tiles staged in bf16 per-lane fragment order with single v8bf (16B)
// ds_store per thread per tile; branchless loads (clamped OOB columns feed
// only never-stored outputs); block-uniform epilogue bounds check.
// =====================================================================
__global__ __launch_bounds__(256) void wmma_gemm_kernel(
    const float* __restrict__ A, const float* __restrict__ Bm,
    float* __restrict__ C, int M, int N, int K) {
  __shared__ v16bf sAf[4][32];   // [16-row tile][lane]
  __shared__ v16bf sBf[4][32];   // [16-col tile][lane]
  const int tid  = threadIdx.x;
  const int lane = tid & 31;
  const int wv   = tid >> 5;
  const int wm   = wv >> 1;          // 0..3
  const int wn   = wv & 1;           // 0..1
  const int m0   = blockIdx.y * 64;
  const int n0   = blockIdx.x * 64;
  const int mm   = lane & 15, half = lane >> 4;

  // A staging: thread -> (row ar, k-octet akb); one 16B LDS store
  const int ar  = tid >> 2, akb = (tid & 3) * 8;
  __bf16* adst = (__bf16*)&sAf[ar >> 4][0] +
                 ((ar & 15) + ((akb >> 4) & 1) * 16) * 16 + (akb & 15);
  // B staging: thread -> (col bn, k-octet bkb); one 16B LDS store
  const int bn  = tid & 63, bkb = (tid >> 6) * 8;
  const int bnc = (n0 + bn < N) ? (n0 + bn) : (N - 1);   // branchless clamp
  __bf16* bdst = (__bf16*)&sBf[bn >> 4][0] +
                 ((bn & 15) + ((bkb >> 4) & 1) * 16) * 16 + (bkb & 15);

  v8f acc0 = {}; v8f acc1 = {};

  for (int k0 = 0; k0 < K; k0 += 32) {
    { // stage A 64x32 (2x global b128 -> 1x ds b128)
      const float* src = A + (size_t)(m0 + ar) * K + k0 + akb;
      float4 u0 = ((const float4*)src)[0];
      float4 u1 = ((const float4*)src)[1];
      float t8[8] = {u0.x, u0.y, u0.z, u0.w, u1.x, u1.y, u1.z, u1.w};
      v8bf pk;
#pragma unroll
      for (int j = 0; j < 8; ++j) pk[j] = (__bf16)t8[j];
      *(v8bf*)adst = pk;
      if (k0 + 32 < K) __builtin_prefetch(src + 32, 0, 3);
    }
    { // stage B 32x64 (8x coalesced b32 -> 1x ds b128)
      const float* src = Bm + (size_t)(k0 + bkb) * N + bnc;
      v8bf pk;
#pragma unroll
      for (int j = 0; j < 8; ++j) pk[j] = (__bf16)src[(size_t)j * N];
      *(v8bf*)bdst = pk;
      if (k0 + 32 < K) __builtin_prefetch(src + (size_t)32 * N, 0, 3);
    }
    __syncthreads();
    v16bf af = sAf[wm][lane];
    acc0 = WMMA_BF16(af, sBf[wn * 2 + 0][lane], acc0);
    acc1 = WMMA_BF16(af, sBf[wn * 2 + 1][lane], acc1);
    __syncthreads();
  }
  // epilogue: block-uniform bounds, pointer-increment stores
  const int gn = n0 + wn * 32 + mm;
  float* cp = C + (size_t)(m0 + wm * 16 + half * 8) * N + gn;
  if (n0 + 64 <= N) {
#pragma unroll
    for (int r = 0; r < 8; ++r) { cp[0] = acc0[r]; cp[16] = acc1[r]; cp += N; }
  } else {
    const bool ok0 = gn < N, ok1 = gn + 16 < N;
#pragma unroll
    for (int r = 0; r < 8; ++r) {
      if (ok0) cp[0]  = acc0[r];
      if (ok1) cp[16] = acc1[r];
      cp += N;
    }
  }
}

// =====================================================================
// 2. causal depthwise conv + SiLU + softplus(dt)+dA (block = token).
// =====================================================================
__global__ __launch_bounds__(256) void conv_dt_kernel(
    const float* __restrict__ zxbcdt, const float* __restrict__ conv_w,
    const float* __restrict__ conv_b, const float* __restrict__ dt_bias,
    const float* __restrict__ A_log, float* __restrict__ xbc,
    float* __restrict__ dt, float* __restrict__ dA) {
  const int bt = blockIdx.x;
  const int b = bt / T_, t = bt % T_;
  const size_t row = (size_t)bt * DPROJ + DIN;
  if (t >= DCONV - 1) {                       // fast path: no boundary
    for (int c = threadIdx.x; c < DXBC; c += 256) {
      float acc = conv_b[c];
#pragma unroll
      for (int k = 0; k < DCONV; ++k)
        acc += zxbcdt[row + (size_t)(k - (DCONV - 1)) * DPROJ + c] * conv_w[k * DXBC + c];
      acc = acc / (1.f + __expf(-acc));
      xbc[(size_t)bt * DXBC + c] = acc;
    }
  } else {
    for (int c = threadIdx.x; c < DXBC; c += 256) {
      float acc = conv_b[c];
#pragma unroll
      for (int k = 0; k < DCONV; ++k) {
        int ts = t + k - (DCONV - 1);
        float v = (ts >= 0) ? zxbcdt[((size_t)(b * T_ + ts) * DPROJ) + DIN + c] : 0.f;
        acc += v * conv_w[k * DXBC + c];
      }
      acc = acc / (1.f + __expf(-acc));
      xbc[(size_t)bt * DXBC + c] = acc;
    }
  }
  if (threadIdx.x < NH) {
    int h = threadIdx.x;
    float raw = zxbcdt[(size_t)bt * DPROJ + DIN + DXBC + h] + dt_bias[h];
    float dtv = (raw > 20.f) ? raw : log1pf(__expf(raw));
    dt[(size_t)bt * NH + h] = dtv;
    dA[(size_t)bt * NH + h] = dtv * (-__expf(A_log[h]));
  }
}

// =====================================================================
// 3. inclusive cumsum of dA within each chunk (block = (b,c,h))
// =====================================================================
__global__ __launch_bounds__(256) void cumsum_kernel(
    const float* __restrict__ dA, float* __restrict__ Acs,
    float* __restrict__ dAtot) {
  const int idx = blockIdx.x;
  const int h = idx % NH;
  const int bc = idx / NH;
  const int l = threadIdx.x;
  __shared__ float buf[CHK];
  buf[l] = dA[((size_t)bc * CHK + l) * NH + h];
  __syncthreads();
  for (int off = 1; off < CHK; off <<= 1) {
    float v = (l >= off) ? buf[l - off] : 0.f;
    __syncthreads();
    buf[l] += v;
    __syncthreads();
  }
  Acs[(size_t)idx * CHK + l] = buf[l];
  if (l == CHK - 1) dAtot[idx] = __expf(buf[l]);
}

// =====================================================================
// 4. CB[l,s] = sum_n C[l,n]*B[s,n]  (lower-triangle tiles), WMMA.
// =====================================================================
__global__ __launch_bounds__(256) void cb_kernel(
    const float* __restrict__ xbc, float* __restrict__ CB) {
  const int bc = blockIdx.x;
  const int lane = threadIdx.x & 31, wv = threadIdx.x >> 5;
  const int mm = lane & 15, half = lane >> 4;
  const float* Cp = xbc + (size_t)bc * CHK * DXBC + DIN + DST;
  const float* Bp = xbc + (size_t)bc * CHK * DXBC + DIN;
  float* out = CB + (size_t)bc * CHK * CHK;
  for (int lti = 0; lti < 2; ++lti) {
    const int lt = wv + lti * 8;
    const float* crow = Cp + (size_t)(lt * 16 + mm) * DXBC;
    for (int st = 0; st <= lt; ++st) {
      const float* brow = Bp + (size_t)(st * 16 + mm) * DXBC;
      v8f acc = {};
#pragma unroll
      for (int ks = 0; ks < 2; ++ks) {
        v16bf af = cvt16(crow + ks * 32 + half * 16);
        v16bf bf = cvt16(brow + ks * 32 + half * 16);
        acc = WMMA_BF16(af, bf, acc);
      }
      float* op = out + (size_t)(lt * 16 + half * 8) * CHK + st * 16 + mm;
#pragma unroll
      for (int r = 0; r < 8; ++r) { op[0] = acc[r]; op += CHK; }
    }
  }
}

// =====================================================================
// 5. per (b,c,h): Y_diag = (CB o L) @ Xdt (+ D*xh), states = Xdt^T @ (decay o B)
// Xdt staged TRANSPOSED in bf16 LDS so both uses are contiguous ds_load_b128.
// =====================================================================
__global__ __launch_bounds__(256) void ydiag_states_kernel(
    const float* __restrict__ xbc, const float* __restrict__ dt,
    const float* __restrict__ Acs, const float* __restrict__ CBbuf,
    const float* __restrict__ Dp, float* __restrict__ Y,
    float* __restrict__ states) {
  const int idx = blockIdx.x;                 // (b*NC+c)*NH + h
  const int h = idx % NH;
  const int bc = idx / NH;
  __shared__ __bf16 sXT[HD][CHK + XPAD];      // Xdt^T: [p][l]
  __shared__ float  sAcs[CHK];
  __shared__ float  sDecay[CHK];
  const int tid = threadIdx.x;

  for (int i = tid; i < CHK; i += 256) sAcs[i] = Acs[(size_t)idx * CHK + i];
  __syncthreads();
  const float alast = sAcs[CHK - 1];
  for (int i = tid; i < CHK; i += 256) sDecay[i] = __expf(alast - sAcs[i]);
  for (int i = tid; i < CHK * HD; i += 256) {
    int l = i >> 6, p = i & 63;               // consecutive tid -> consecutive p
    size_t bt = (size_t)bc * CHK + l;
    sXT[p][l] = (__bf16)(xbc[bt * DXBC + h * HD + p] * dt[bt * NH + h]);
  }
  __syncthreads();

  const int lane = tid & 31, wv = tid >> 5;
  const int mm = lane & 15, half = lane >> 4;
  const float* cb = CBbuf + (size_t)bc * CHK * CHK;
  const float dh = Dp[h];

  // ---- phase A: Y_diag
  for (int lti = 0; lti < 2; ++lti) {
    const int lt = wv + lti * 8;
    const int l = lt * 16 + mm;
    const float al = sAcs[l];
    const float* crow = cb + (size_t)l * CHK;
    v8f acc[4] = {};
    const int smax = lt * 16 + 15;
    for (int sb = 0; sb * 32 <= smax; ++sb) {
      const int s0 = sb * 32 + half * 16, s1 = sb * 32 + half * 16 + 8;
      float tr[16];
      *(float4*)(tr + 0)  = ((const float4*)(crow + s0))[0];
      *(float4*)(tr + 4)  = ((const float4*)(crow + s0))[1];
      *(float4*)(tr + 8)  = ((const float4*)(crow + s1))[0];
      *(float4*)(tr + 12) = ((const float4*)(crow + s1))[1];
      v16bf af;
#pragma unroll
      for (int e = 0; e < 16; ++e) {
        int s = (e < 8) ? (s0 + e) : (s1 + e - 8);
        af[e] = (__bf16)((s <= l) ? tr[e] * __expf(al - sAcs[s]) : 0.f);
      }
#pragma unroll
      for (int pt = 0; pt < 4; ++pt) {
        v16bf bfv = *(const v16bf*)&sXT[pt * 16 + mm][sb * 32 + half * 16];
        acc[pt] = WMMA_BF16(af, bfv, acc[pt]);
      }
    }
    const float* xp = xbc + ((size_t)bc * CHK + lt * 16 + half * 8) * DXBC + h * HD + mm;
    float*       yp = Y   + ((size_t)bc * CHK + lt * 16 + half * 8) * DIN  + h * HD + mm;
#pragma unroll
    for (int r = 0; r < 8; ++r) {
#pragma unroll
      for (int pt = 0; pt < 4; ++pt) yp[pt * 16] = acc[pt][r] + dh * xp[pt * 16];
      xp += DXBC; yp += DIN;
    }
  }

  // ---- phase B: states[p,n] = sum_l Xdt[l,p]*decay[l]*B[l,n]
  {
    const int pt = wv & 3;
    const int ntb = (wv >> 2) * 2;
    const int p = pt * 16 + mm;
    const float* Bp = xbc + (size_t)bc * CHK * DXBC + DIN;
    v8f acc[2] = {};
    for (int kb = 0; kb < 8; ++kb) {
      v16bf af = lds_runs(&sXT[p][kb * 32 + half * 16],
                          &sXT[p][kb * 32 + half * 16 + 8]);
      const int l0 = kb * 32 + half * 16;
#pragma unroll
      for (int j = 0; j < 2; ++j) {
        const float* bp = Bp + (size_t)l0 * DXBC + (ntb + j) * 16 + mm;
        v16bf bfv;
#pragma unroll
        for (int e = 0; e < 16; ++e)
          bfv[e] = (__bf16)(sDecay[l0 + e] * bp[(size_t)e * DXBC]);
        acc[j] = WMMA_BF16(af, bfv, acc[j]);
      }
    }
    float* sp = states + (size_t)idx * (HD * DST) +
                (pt * 16 + half * 8) * DST + ntb * 16 + mm;
#pragma unroll
    for (int r = 0; r < 8; ++r) {
      sp[0]  = acc[0][r];
      sp[16] = acc[1][r];
      sp += DST;
    }
  }
}

// =====================================================================
// 6. sequential inter-chunk scan per (b,h): Y += sd o (C @ h^T); h = h*dtot + st
// =====================================================================
__global__ __launch_bounds__(256) void yoff_kernel(
    const float* __restrict__ xbc, const float* __restrict__ Acs,
    const float* __restrict__ dAtot, const float* __restrict__ states,
    float* __restrict__ Y) {
  const int b = blockIdx.x / NH, h = blockIdx.x % NH;
  __shared__ float sh[HD * DST];               // running state h[p][n]
  __shared__ float sSd[CHK];                   // exp(Acs) for current chunk
  const int tid = threadIdx.x;
  for (int i = tid; i < HD * DST; i += 256) sh[i] = 0.f;
  const int lane = tid & 31, wv = tid >> 5;
  const int mm = lane & 15, half = lane >> 4;

  for (int c = 0; c < NC; ++c) {
    const int bc = b * NC + c;
    const int idx = bc * NH + h;
    for (int i = tid; i < CHK; i += 256)
      sSd[i] = __expf(Acs[(size_t)idx * CHK + i]);
    __syncthreads();
    const float* Cp = xbc + (size_t)bc * CHK * DXBC + DIN + DST;
    for (int lti = 0; lti < 2; ++lti) {
      const int lt = wv + lti * 8;
      const float* crow = Cp + (size_t)(lt * 16 + mm) * DXBC;
      v8f acc[4] = {};
#pragma unroll
      for (int kb = 0; kb < 2; ++kb) {
        v16bf af = cvt16(crow + kb * 32 + half * 16);
#pragma unroll
        for (int pt = 0; pt < 4; ++pt) {
          v16bf bfv = cvt16(&sh[(pt * 16 + mm) * DST + kb * 32 + half * 16]);
          acc[pt] = WMMA_BF16(af, bfv, acc[pt]);
        }
      }
      float* yp = Y + ((size_t)bc * CHK + lt * 16 + half * 8) * DIN + h * HD + mm;
      const float* sdp = &sSd[lt * 16 + half * 8];
#pragma unroll
      for (int r = 0; r < 8; ++r) {
        const float sd = sdp[r];
#pragma unroll
        for (int pt = 0; pt < 4; ++pt) yp[pt * 16] += acc[pt][r] * sd;
        yp += DIN;
      }
    }
    __syncthreads();                            // all reads of sh/sSd done
    const float dtot = dAtot[idx];
    const float* st = states + (size_t)idx * (HD * DST);
    for (int i = tid; i < HD * DST; i += 256) sh[i] = sh[i] * dtot + st[i];
    __syncthreads();
  }
}

// =====================================================================
// 7. gating y*silu(z) + RMSNorm(norm_w), in-place on Y. block = token.
// =====================================================================
__global__ __launch_bounds__(256) void gate_norm_kernel(
    const float* __restrict__ zxbcdt, const float* __restrict__ norm_w,
    float* __restrict__ Y) {
  const int bt = blockIdx.x;
  __shared__ float red[256];
  float vals[4];
  float ss = 0.f;
#pragma unroll
  for (int j = 0; j < 4; ++j) {
    int i = threadIdx.x + j * 256;
    float y = Y[(size_t)bt * DIN + i];
    float z = zxbcdt[(size_t)bt * DPROJ + i];
    y *= z / (1.f + __expf(-z));
    vals[j] = y;
    ss += y * y;
  }
  red[threadIdx.x] = ss;
  __syncthreads();
  for (int off = 128; off > 0; off >>= 1) {
    if (threadIdx.x < off) red[threadIdx.x] += red[threadIdx.x + off];
    __syncthreads();
  }
  const float scale = rsqrtf(red[0] / (float)DIN + 1e-5f);
#pragma unroll
  for (int j = 0; j < 4; ++j) {
    int i = threadIdx.x + j * 256;
    Y[(size_t)bt * DIN + i] = vals[j] * scale * norm_w[i];
  }
}

// =====================================================================
extern "C" void kernel_launch(void* const* d_in, const int* in_sizes, int n_in,
                              void* d_out, int out_size, void* d_ws, size_t ws_size,
                              hipStream_t stream) {
  (void)in_sizes; (void)n_in; (void)out_size; (void)ws_size;
  const float* x       = (const float*)d_in[0];
  const float* W_in    = (const float*)d_in[1];
  const float* conv_w  = (const float*)d_in[2];
  const float* conv_b  = (const float*)d_in[3];
  const float* dt_bias = (const float*)d_in[4];
  const float* A_log   = (const float*)d_in[5];
  const float* Dp      = (const float*)d_in[6];
  const float* norm_w  = (const float*)d_in[7];
  const float* W_out   = (const float*)d_in[8];
  float* out = (float*)d_out;

  // workspace layout (floats); total ~162 MB
  float* ws      = (float*)d_ws;
  float* zxbcdt  = ws;                                    // 8192*2192
  float* xbcB    = zxbcdt + (size_t)8192 * DPROJ;         // 8192*1152
  float* dtB     = xbcB   + (size_t)8192 * DXBC;          // 8192*16
  float* dAB     = dtB    + (size_t)8192 * NH;            // 8192*16
  float* AcsB    = dAB    + (size_t)8192 * NH;            // 512*256
  float* dAtotB  = AcsB   + (size_t)512 * CHK;            // 512
  float* CBB     = dAtotB + 512;                          // 32*256*256
  float* statesB = CBB    + (size_t)B_ * NC * CHK * CHK;  // 512*64*64
  float* YB      = statesB + (size_t)512 * HD * DST;      // 8192*1024

  const int BT = B_ * T_;                                 // 8192

  wmma_gemm_kernel<<<dim3((DPROJ + 63) / 64, BT / 64), 256, 0, stream>>>(
      x, W_in, zxbcdt, BT, DPROJ, DM);
  conv_dt_kernel<<<BT, 256, 0, stream>>>(zxbcdt, conv_w, conv_b, dt_bias,
                                         A_log, xbcB, dtB, dAB);
  cumsum_kernel<<<B_ * NC * NH, 256, 0, stream>>>(dAB, AcsB, dAtotB);
  cb_kernel<<<B_ * NC, 256, 0, stream>>>(xbcB, CBB);
  ydiag_states_kernel<<<B_ * NC * NH, 256, 0, stream>>>(
      xbcB, dtB, AcsB, CBB, Dp, YB, statesB);
  yoff_kernel<<<B_ * NH, 256, 0, stream>>>(xbcB, AcsB, dAtotB, statesB, YB);
  gate_norm_kernel<<<BT, 256, 0, stream>>>(zxbcdt, norm_w, YB);
  wmma_gemm_kernel<<<dim3(DM / 64, BT / 64), 256, 0, stream>>>(
      YB, W_out, out, BT, DM, DIN);
}
